// position_transform_23278722744765
// MI455X (gfx1250) — matrix operand order repaired and checked by reading, not compiled, not run on hardware
//
#include <hip/hip_runtime.h>
#include <hip/hip_bf16.h>
#include <stdint.h>

// ---------------------------------------------------------------------------
// Problem constants (from reference)
// ---------------------------------------------------------------------------
constexpr int Bn  = 65536;
constexpr int Dn  = 1024;
constexpr int H1n = 512;
constexpr int H2n = 256;
constexpr float EPS = 1e-5f;

// CDNA5 WMMA vector types (wave32)
typedef __attribute__((ext_vector_type(16))) __bf16 v16bf;
typedef __attribute__((ext_vector_type(8)))  float  v8f;

// GEMM tiling: 128x64 tile per 256-thread block (8 waves), K step = 32
#define TM   128
#define TN   64
#define TK   32
#define LDSK 40   // padded K-stride in halfwords (80B: 16B-aligned, bank-skewed)

// Exactly one v_cvt_pk_bf16_f32: dst[15:0]=cvt(a), dst[31:16]=cvt(b).
// (The builtin is absent on this toolchain and the scalar-cast fallback makes
// clang re-pack halves with a vALU storm, so issue the instruction directly.)
__device__ __forceinline__ unsigned pack_bf16(float a, float b) {
  unsigned r;
  asm("v_cvt_pk_bf16_f32 %0, %1, %2" : "=v"(r) : "v"(a), "v"(b));
  return r;
}

__device__ __forceinline__ void wait_async_all() {
#if __has_builtin(__builtin_amdgcn_s_wait_asynccnt)
  __builtin_amdgcn_s_wait_asynccnt(0);
#else
  asm volatile("s_wait_asynccnt 0x0" ::: "memory");
#endif
}

// Async global -> LDS copy of one 16-byte chunk (ASYNCcnt-tracked, per-lane)
__device__ __forceinline__ void async_copy_b128(unsigned ldsOff,
                                                const void* gptr) {
  asm volatile("global_load_async_to_lds_b128 %0, %1, off"
               :: "v"(ldsOff),
                  "v"((unsigned long long)(uintptr_t)gptr)
               : "memory");
}

// ---------------------------------------------------------------------------
// fp32 [K][N] -> bf16 transposed [N][K] weight conversion (tiny matrices)
// ---------------------------------------------------------------------------
__global__ void pt_cvt_wT(const float* __restrict__ in,
                          unsigned short* __restrict__ out, int K, int N) {
  int i = blockIdx.x * 256 + threadIdx.x;
  if (i < K * N) {
    int k = i / N, n = i - k * N;
    out[(size_t)n * K + k] = (unsigned short)(pack_bf16(in[i], 0.f) & 0xffffu);
  }
}

__global__ void pt_zero(float* __restrict__ p, int n) {
  int i = blockIdx.x * 256 + threadIdx.x;
  if (i < n) p[i] = 0.f;
}

// ---------------------------------------------------------------------------
// Tiled bf16 WMMA GEMM:  Hout = op(A) @ W + bias, plus per-column batch stats.
//   op(A) = BN_A ? relu(A*sc + sh) : A     (fused into the bf16 conversion)
// A:  [Bn, KT] fp32 row-major.
// Wt: [N, KT] bf16 (pre-transposed) -> B tiles async-copied straight to LDS.
// Double-buffered K loop: next tile's global A loads + async B copy issue
// before the current tile's WMMAs; A cvt+LDS stores land after them.
// ---------------------------------------------------------------------------
template<int KT, int N, bool BN_A>
__global__ __launch_bounds__(256)
void pt_gemm_bn(const float* __restrict__ A,
                const unsigned short* __restrict__ Wt,
                const float* __restrict__ bias,
                const float* __restrict__ scA,
                const float* __restrict__ shA,
                float* __restrict__ Hout,
                float* __restrict__ gsum,
                float* __restrict__ gsq) {
  __shared__ __align__(16) unsigned short sA[2][TM * LDSK]; // [m][k]
  __shared__ __align__(16) unsigned short sB[2][TN * LDSK]; // [n][k]
  __shared__ float cSum[TN];
  __shared__ float cSq[TN];
  __shared__ float sSc[BN_A ? KT : 1];
  __shared__ float sSh[BN_A ? KT : 1];

  const int tid  = threadIdx.x;
  const int lane = tid & 31;
  const int wv   = tid >> 5;
  const int wr   = wv >> 1;      // wave row group 0..3 (32 rows each)
  const int wc   = wv & 1;       // wave col group 0..1 (32 cols each)
  const int hh   = lane >> 4;    // lane half
  const int l    = lane & 15;

  const int row0 = blockIdx.x * TM;
  const int col0 = blockIdx.y * TN;

  if (tid < TN) { cSum[tid] = 0.f; cSq[tid] = 0.f; }
  if (BN_A) {
    for (int i = tid; i < KT; i += 256) { sSc[i] = scA[i]; sSh[i] = shA[i]; }
  }

  // ---- staging helpers -----------------------------------------------------
  const int ar  = tid >> 3;        // A row handled by this thread (q stride 32)
  const int ac4 = tid & 7;         // A float4 column within row
  auto load_A = [&](int kk, float4* av) {
    #pragma unroll
    for (int q = 0; q < 4; ++q)
      av[q] = *(const float4*)(A + (size_t)(row0 + ar + 32 * q) * KT +
                               kk + ac4 * 4);
  };
  auto store_A = [&](int bsel, int kk, const float4* av) {
    #pragma unroll
    for (int q = 0; q < 4; ++q) {
      float x0 = av[q].x, x1 = av[q].y, x2 = av[q].z, x3 = av[q].w;
      if (BN_A) {
        int k = kk + ac4 * 4;
        x0 = fmaxf(0.f, fmaf(x0, sSc[k + 0], sSh[k + 0]));
        x1 = fmaxf(0.f, fmaf(x1, sSc[k + 1], sSh[k + 1]));
        x2 = fmaxf(0.f, fmaf(x2, sSc[k + 2], sSh[k + 2]));
        x3 = fmaxf(0.f, fmaf(x3, sSc[k + 3], sSh[k + 3]));
      }
      *(uint2*)&sA[bsel][(ar + 32 * q) * LDSK + ac4 * 4] =
          make_uint2(pack_bf16(x0, x1), pack_bf16(x2, x3));
    }
  };
  const int bn = tid >> 2;         // B row (tile col) 0..63
  const int bc = tid & 3;          // 16-byte chunk within row
  auto issue_B = [&](int bsel, int kk) {
    unsigned ldsOff =
        (unsigned)(uintptr_t)&sB[bsel][bn * LDSK] + bc * 16;
    const char* gp =
        (const char*)(Wt + (size_t)(col0 + bn) * KT + kk) + bc * 16;
    async_copy_b128(ldsOff, gp);   // global_load_async_to_lds_b128
  };

  // ---- prologue: stage tile 0 ---------------------------------------------
  {
    float4 av0[4];
    load_A(0, av0);
    issue_B(0, 0);
    store_A(0, 0, av0);
  }
  wait_async_all();
  __syncthreads();

  v8f acc[2][2] = {};
  int buf = 0;

  for (int k0 = 0; k0 < KT; k0 += TK) {
    const int  nxt  = buf ^ 1;
    const bool more = (k0 + TK) < KT;   // uniform across block

    float4 av[4];
    if (more) {            // issue next tile's global traffic first
      load_A(k0 + TK, av);
      issue_B(nxt, k0 + TK);
    }

    // ---- fragments: contiguous k = 16*hh + s, same order for A and B ----
    union FR { uint4 q[2]; v16bf v; } fa[2], fb[2];
    #pragma unroll
    for (int mt = 0; mt < 2; ++mt) {
      int m = wr * 32 + mt * 16 + l;
      fa[mt].q[0] = *(const uint4*)&sA[buf][m * LDSK + 16 * hh];
      fa[mt].q[1] = *(const uint4*)&sA[buf][m * LDSK + 16 * hh + 8];
    }
    #pragma unroll
    for (int nt = 0; nt < 2; ++nt) {
      int n = wc * 32 + nt * 16 + l;
      fb[nt].q[0] = *(const uint4*)&sB[buf][n * LDSK + 16 * hh];
      fb[nt].q[1] = *(const uint4*)&sB[buf][n * LDSK + 16 * hh + 8];
    }
    #pragma unroll
    for (int mt = 0; mt < 2; ++mt)
      #pragma unroll
      for (int nt = 0; nt < 2; ++nt)
        acc[mt][nt] = __builtin_amdgcn_wmma_f32_16x16x32_bf16(
            false, fa[mt].v, false, fb[nt].v,
            (short)0, acc[mt][nt], false, false);

    if (more) store_A(nxt, k0 + TK, av);  // cvt lands after the WMMAs

    wait_async_all();
    __syncthreads();
    buf = nxt;
  }

  // ---- epilogue: bias add, store, batch-stat accumulation ----
  float lsum[2] = {0.f, 0.f}, lsq[2] = {0.f, 0.f};
  #pragma unroll
  for (int nt = 0; nt < 2; ++nt) {
    int col = col0 + wc * 32 + nt * 16 + l;
    float bv = bias[col];
    #pragma unroll
    for (int mt = 0; mt < 2; ++mt) {
      #pragma unroll
      for (int e = 0; e < 8; ++e) {
        int row = row0 + wr * 32 + mt * 16 + hh * 8 + e;  // D-matrix layout
        float v = acc[mt][nt][e] + bv;
        Hout[(size_t)row * N + col] = v;
        lsum[nt] += v;
        lsq[nt]  += v * v;
      }
    }
  }
  #pragma unroll
  for (int nt = 0; nt < 2; ++nt) {
    int c = wc * 32 + nt * 16 + l;
    atomicAdd(&cSum[c], lsum[nt]);   // ds_add_f32
    atomicAdd(&cSq[c],  lsq[nt]);
  }
  __syncthreads();
  if (tid < TN) {
    atomicAdd(&gsum[col0 + tid], cSum[tid]);  // global_atomic_add_f32
    atomicAdd(&gsq[col0 + tid],  cSq[tid]);
  }
}

// ---------------------------------------------------------------------------
// BN finalize: fold mean/var + gamma/beta into per-column scale & shift
// ---------------------------------------------------------------------------
__global__ void pt_bn_finalize(const float* __restrict__ gsum,
                               const float* __restrict__ gsq,
                               const float* __restrict__ g,
                               const float* __restrict__ be,
                               float* __restrict__ sc,
                               float* __restrict__ sh,
                               int n, float invB) {
  int i = blockIdx.x * 256 + threadIdx.x;
  if (i < n) {
    float mu  = gsum[i] * invB;
    float var = fmaxf(gsq[i] * invB - mu * mu, 0.f);
    float rs  = rsqrtf(var + EPS);
    float s   = rs * g[i];
    sc[i] = s;
    sh[i] = be[i] - mu * s;
  }
}

// ---------------------------------------------------------------------------
// Head: BN+ReLU on h2, y = h@W3 + b3 (N=12), then SO(3) projection (Horn's
// quaternion method: dominant eigenvector of the 4x4 K matrix) + translation.
// One wave32 per row; coalesced h2 reads; shuffle reduction.
// ---------------------------------------------------------------------------
__global__ __launch_bounds__(256)
void pt_head(const float* __restrict__ H2,
             const float* __restrict__ W3,   // [256,12]
             const float* __restrict__ b3,   // [12]
             const float* __restrict__ sc2,
             const float* __restrict__ sh2,
             float* __restrict__ out) {      // [B,3,4]
  __shared__ float sW[H2n * 12];
  __shared__ float sb[12];
  __shared__ float ssc[H2n];
  __shared__ float ssh[H2n];

  const int tid = threadIdx.x;
  for (int i = tid; i < H2n * 12; i += 256) sW[i] = W3[i];
  if (tid < 12) sb[tid] = b3[tid];
  if (tid < H2n) { ssc[tid] = sc2[tid]; ssh[tid] = sh2[tid]; }
  __syncthreads();

  const int lane = tid & 31;
  const int wv   = tid >> 5;
  const size_t row = (size_t)blockIdx.x * 8 + wv;

  const float* hr = H2 + row * H2n;
  float acc[12];
  #pragma unroll
  for (int c = 0; c < 12; ++c) acc[c] = 0.f;

  #pragma unroll
  for (int j = 0; j < 8; ++j) {
    int k = lane + 32 * j;
    float v = fmaxf(0.f, fmaf(hr[k], ssc[k], ssh[k]));
    #pragma unroll
    for (int c = 0; c < 12; ++c) acc[c] = fmaf(v, sW[k * 12 + c], acc[c]);
  }
  #pragma unroll
  for (int off = 16; off > 0; off >>= 1)
    #pragma unroll
    for (int c = 0; c < 12; ++c)
      acc[c] += __shfl_xor(acc[c], off, 32);

  if (lane == 0) {
    float y[12];
    #pragma unroll
    for (int c = 0; c < 12; ++c) y[c] = acc[c] + sb[c];

    const float Sxx = y[0], Sxy = y[1], Sxz = y[2];
    const float Syx = y[3], Syy = y[4], Syz = y[5];
    const float Szx = y[6], Szy = y[7], Szz = y[8];

    float K[4][4];
    K[0][0] =  Sxx + Syy + Szz; K[0][1] = Syz - Szy;       K[0][2] = Szx - Sxz;        K[0][3] = Sxy - Syx;
    K[1][0] =  K[0][1];         K[1][1] = Sxx - Syy - Szz; K[1][2] = Sxy + Syx;        K[1][3] = Szx + Sxz;
    K[2][0] =  K[0][2];         K[2][1] = K[1][2];         K[2][2] = -Sxx + Syy - Szz; K[2][3] = Syz + Szy;
    K[3][0] =  K[0][3];         K[3][1] = K[1][3];         K[3][2] = K[2][3];          K[3][3] = -Sxx - Syy + Szz;

    float fn = sqrtf(Sxx*Sxx + Sxy*Sxy + Sxz*Sxz + Syx*Syx + Syy*Syy +
                     Syz*Syz + Szx*Szx + Szy*Szy + Szz*Szz) + 1e-12f;
    float shift = 2.f * fn;            // make dominant eigenvalue positive
    K[0][0] += shift; K[1][1] += shift; K[2][2] += shift; K[3][3] += shift;

    float q0 = 1.f, q1 = 0.f, q2 = 0.f, q3 = 0.f;
    #pragma unroll 1
    for (int it = 0; it < 16; ++it) {  // power iteration -> dominant eigvec
      float t0 = K[0][0]*q0 + K[0][1]*q1 + K[0][2]*q2 + K[0][3]*q3;
      float t1 = K[1][0]*q0 + K[1][1]*q1 + K[1][2]*q2 + K[1][3]*q3;
      float t2 = K[2][0]*q0 + K[2][1]*q1 + K[2][2]*q2 + K[2][3]*q3;
      float t3 = K[3][0]*q0 + K[3][1]*q1 + K[3][2]*q2 + K[3][3]*q3;
      float nr = rsqrtf(t0*t0 + t1*t1 + t2*t2 + t3*t3 + 1e-30f);
      q0 = t0 * nr; q1 = t1 * nr; q2 = t2 * nr; q3 = t3 * nr;
    }
    const float w = q0, xq = q1, yq = q2, zq = q3;
    float R[3][3];
    R[0][0] = 1.f - 2.f*(yq*yq + zq*zq); R[0][1] = 2.f*(xq*yq - w*zq);        R[0][2] = 2.f*(xq*zq + w*yq);
    R[1][0] = 2.f*(xq*yq + w*zq);        R[1][1] = 1.f - 2.f*(xq*xq + zq*zq); R[1][2] = 2.f*(yq*zq - w*xq);
    R[2][0] = 2.f*(xq*zq - w*yq);        R[2][1] = 2.f*(yq*zq + w*xq);        R[2][2] = 1.f - 2.f*(xq*xq + yq*yq);

    float* o = out + row * 12;
    o[0]  = R[0][0]; o[1]  = R[0][1]; o[2]  = R[0][2]; o[3]  = y[9];
    o[4]  = R[1][0]; o[5]  = R[1][1]; o[6]  = R[1][2]; o[7]  = y[10];
    o[8]  = R[2][0]; o[9]  = R[2][1]; o[10] = R[2][2]; o[11] = y[11];
  }
}

// ---------------------------------------------------------------------------
// Launcher
// ---------------------------------------------------------------------------
extern "C" void kernel_launch(void* const* d_in, const int* in_sizes, int n_in,
                              void* d_out, int out_size, void* d_ws, size_t ws_size,
                              hipStream_t stream) {
  const float* x   = (const float*)d_in[0];
  const float* W1  = (const float*)d_in[1];
  const float* b1  = (const float*)d_in[2];
  const float* g1  = (const float*)d_in[3];
  const float* be1 = (const float*)d_in[4];
  const float* W2  = (const float*)d_in[5];
  const float* b2  = (const float*)d_in[6];
  const float* g2  = (const float*)d_in[7];
  const float* be2 = (const float*)d_in[8];
  const float* W3  = (const float*)d_in[9];
  const float* b3  = (const float*)d_in[10];
  float* out = (float*)d_out;

  // Workspace layout
  float* h1 = (float*)d_ws;                                   // B*H1 fp32
  float* h2 = h1 + (size_t)Bn * H1n;                          // B*H2 fp32
  unsigned short* w1t = (unsigned short*)(h2 + (size_t)Bn * H2n); // [H1][D] bf16
  unsigned short* w2t = w1t + (size_t)Dn * H1n;               // [H2][H1] bf16
  float* stats = (float*)(w2t + (size_t)H1n * H2n);
  float* gsum1 = stats;             // 512
  float* gsq1  = stats + 512;       // 512
  float* gsum2 = stats + 1024;      // 256
  float* gsq2  = stats + 1280;      // 256  (first 1536 floats zeroed per call)
  float* sc1   = stats + 1536;      // 512
  float* sh1   = stats + 2048;      // 512
  float* sc2   = stats + 2560;      // 256
  float* sh2   = stats + 2816;      // 256

  const float invB = 1.0f / (float)Bn;

  pt_zero<<<(1536 + 255) / 256, 256, 0, stream>>>(stats, 1536);
  pt_cvt_wT<<<(Dn * H1n) / 256, 256, 0, stream>>>(W1, w1t, Dn, H1n);
  pt_cvt_wT<<<(H1n * H2n) / 256, 256, 0, stream>>>(W2, w2t, H1n, H2n);

  // Layer 1: h1 = x @ W1 + b1  (+ batch stats)
  pt_gemm_bn<Dn, H1n, false>
      <<<dim3(Bn / TM, H1n / TN), 256, 0, stream>>>(
          x, w1t, b1, nullptr, nullptr, h1, gsum1, gsq1);
  pt_bn_finalize<<<2, 256, 0, stream>>>(gsum1, gsq1, g1, be1, sc1, sh1,
                                        H1n, invB);

  // Layer 2: h2 = relu(bn(h1)) @ W2 + b2  (+ batch stats)
  pt_gemm_bn<H1n, H2n, true>
      <<<dim3(Bn / TM, H2n / TN), 256, 0, stream>>>(
          h1, w2t, b2, sc1, sh1, h2, gsum2, gsq2);
  pt_bn_finalize<<<1, 256, 0, stream>>>(gsum2, gsq2, g2, be2, sc2, sh2,
                                        H2n, invB);

  // Head: y = relu(bn(h2)) @ W3 + b3, SO(3) projection, pack [R|t]
  pt_head<<<Bn / 8, 256, 0, stream>>>(h2, W3, b3, sc2, sh2, out);
}
